// AttentionBlock_38268158607694
// MI455X (gfx1250) — matrix operand order repaired and checked by reading, not compiled, not run on hardware
//
#include <hip/hip_runtime.h>

// ---------------------------------------------------------------------------
// Talking-heads attention for MI455X (gfx1250, wave32, WMMA bf16 + TDM).
// B=8, S=1024, C=768, H=12, D=64.
// ---------------------------------------------------------------------------

typedef __attribute__((ext_vector_type(16))) __bf16 bf16x16;
typedef __attribute__((ext_vector_type(8)))  __bf16 bf16x8;
typedef __attribute__((ext_vector_type(8)))  float  f32x8;

#define NB 8
#define NS 1024
#define NC 768
#define NH 12
#define ND 64

// ---------------------------------------------------------------------------
// Tensor Data Mover: load a 64x64 bf16 tile (row-major, row stride = ld elems)
// into LDS at lds_off. D# packing per CDNA5 ISA 8.3/8.4 (group0/group1).
// Toolchain uses the 6-arg builtin: (v4u g0, v8i g1, v4i g2, v4i g3, v8i, cpol)
// ---------------------------------------------------------------------------
#if defined(__has_builtin)
#if __has_builtin(__builtin_amdgcn_tensor_load_to_lds)
#define HAVE_TDM 1
#endif
#endif

#ifdef HAVE_TDM
typedef __attribute__((ext_vector_type(4))) unsigned int u32x4;
typedef __attribute__((ext_vector_type(8))) int i32x8;
typedef __attribute__((ext_vector_type(4))) int i32x4;

__device__ __forceinline__ void tdm_load_tile64x64_bf16(
    const __bf16* gptr, unsigned lds_off, unsigned ld /*elems*/)
{
  const unsigned long long ga = (unsigned long long)gptr;
  u32x4 g0;
  g0[0] = 1u;                                   // count=1, user descriptor
  g0[1] = lds_off;                              // LDS byte address
  g0[2] = (unsigned)(ga & 0xffffffffu);         // global_addr[31:0]
  g0[3] = (unsigned)((ga >> 32) & 0x01ffffffu)  // global_addr[56:32]
        | (2u << 30);                           // type=2 ("image")
  const unsigned dim0 = ld;                     // tensor dim0 = row length
  const unsigned dim1 = 1u << 20;               // rows (OOB bound, generous)
  i32x8 g1;
  g1[0] = (int)(1u << 16);                      // data_size=1 (2 bytes)
  g1[1] = (int)((dim0 & 0xffffu) << 16);        // tensor_dim0[15:0]
  g1[2] = (int)(((dim0 >> 16) & 0xffffu) | ((dim1 & 0xffffu) << 16));
  g1[3] = (int)(((dim1 >> 16) & 0xffffu) | (64u << 16));  // tile_dim0=64
  g1[4] = (int)64u;                             // tile_dim1=64, tile_dim2=0
  g1[5] = (int)dim0;                            // tensor_dim0_stride[31:0]
  g1[6] = 0;                                    // stride0[47:32] | stride1 lo
  g1[7] = 0;
  const i32x4 z4 = {0, 0, 0, 0};
  const i32x8 z8 = {0, 0, 0, 0, 0, 0, 0, 0};
  __builtin_amdgcn_tensor_load_to_lds(g0, g1, z4, z4, z8, 0);
}
#endif

// ---- WMMA fragment loaders (CDNA5 ISA 7.12.2 layouts, wave32) -------------
// A strip: 16 rows x 64 cols bf16 in LDS (row stride 64). Lane m = lane&15,
// half = lane>>4. Elements = two contiguous runs of 8: K = kh+h8+{0..7} and
// K = kh+16+h8+{0..7}  -> two b128 LDS loads.
__device__ __forceinline__ bf16x16 frag_a_ld(const __bf16* strip, int kh) {
  const int lane = threadIdx.x & 31;
  const int h8 = (lane >> 4) << 3;
  const __bf16* row = strip + (lane & 15) * 64 + kh + h8;
  const bf16x8 lo = *(const bf16x8*)(row);
  const bf16x8 hi = *(const bf16x8*)(row + 16);
  return __builtin_shufflevector(lo, hi, 0,1,2,3,4,5,6,7,8,9,10,11,12,13,14,15);
}
// B tile stored column-major [n][k] (64 x 64, stride 64). Lane n = lane&15;
// half = lane>>4, h8 = 8*half. Elements = K = kh+h8+{0..7} and
// K = kh+16+h8+{0..7} (same order as A) -> two b128 LDS loads.
__device__ __forceinline__ bf16x16 frag_b_ld(const __bf16* Bs, int n0, int kh) {
  const int lane = threadIdx.x & 31;
  const int h8 = (lane >> 4) << 3;
  const __bf16* p = Bs + (n0 + (lane & 15)) * 64 + kh + h8;
  const bf16x8 lo = *(const bf16x8*)(p);
  const bf16x8 hi = *(const bf16x8*)(p + 16);
  return __builtin_shufflevector(lo, hi, 0,1,2,3,4,5,6,7,8,9,10,11,12,13,14,15);
}

// ---------------------------------------------------------------------------
// Generic 64x64-tile WMMA GEMM, K-step 64, 128 threads (4 waves).
// MODE 0: out bf16 [B,H,S,D]   (QKV projections; m=(b,s), n=(h,d))
// MODE 1: out bf16 [z][m][n]   (raw attention scores, z = b*H+h)
// MODE 2: out bf16 [B,S,H,D]   (attn @ V; z=(b,h), n=d)
// MODE 3: out f32  [m*NC+n]   (final projection -> d_out)
// AF32/BF32: f32 global operand (convert to bf16 while staging)
// TRANSB:   B tile source is [n][k] row-major (K^T case) -> contiguous copy
// ---------------------------------------------------------------------------
template <int MODE, bool AF32, bool BF32, bool TRANSB>
__global__ __launch_bounds__(128)
void gemm_wmma(const void* __restrict__ Ap, const void* __restrict__ Bp,
               void* __restrict__ Op, int lda, int ldb, int ksteps,
               long bsA, long bsB, float bscale)
{
  __shared__ __align__(16) __bf16 As[64 * 64];   // [m][k]
  __shared__ __align__(16) __bf16 Bs[64 * 64];   // [n][k]

  const int t  = threadIdx.x;
  const int m0 = blockIdx.x * 64;
  const int n0 = blockIdx.y * 64;
  const int z  = blockIdx.z;
  const int wv = t >> 5;
  const long zA = (long)z * bsA;
  const long zB = (long)z * bsB;

  f32x8 acc[4] = {};

  for (int kt = 0; kt < ksteps; ++kt) {
    const int k0 = kt * 64;
    bool tdm_used = false;
    // ---- stage A tile [64 rows][64 k] ----
    if (AF32) {
      const int row = t >> 1, c = (t & 1) << 5;
      const float* Af = (const float*)Ap + zA + (long)(m0 + row) * lda + k0 + c;
#pragma unroll
      for (int j = 0; j < 4; ++j) {
        bf16x8 t8;
#pragma unroll
        for (int u = 0; u < 8; ++u) t8[u] = (__bf16)Af[8 * j + u];
        *(bf16x8*)&As[row * 64 + c + 8 * j] = t8;
      }
    } else {
      const __bf16* Ab = (const __bf16*)Ap + zA + (long)m0 * lda + k0;
#ifdef HAVE_TDM
      if (wv == 0)
        tdm_load_tile64x64_bf16(
            Ab, (unsigned)(unsigned long long)(const void*)&As[0], (unsigned)lda);
      tdm_used = true;
#else
      const int row = t >> 1, c = (t & 1) << 5;
      const __bf16* Ar = Ab + (long)row * lda + c;
#pragma unroll
      for (int j = 0; j < 4; ++j)
        *(bf16x8*)&As[row * 64 + c + 8 * j] = *(const bf16x8*)(Ar + 8 * j);
#endif
    }
    // ---- stage B tile into [n][k] ----
    if (TRANSB) {
      const __bf16* Bb = (const __bf16*)Bp + zB + (long)n0 * ldb + k0;
#ifdef HAVE_TDM
      if (wv == 0)
        tdm_load_tile64x64_bf16(
            Bb, (unsigned)(unsigned long long)(const void*)&Bs[0], (unsigned)ldb);
      tdm_used = true;
#else
      const int nr = t >> 1, c = (t & 1) << 5;
      const __bf16* Br = Bb + (long)nr * ldb + c;
#pragma unroll
      for (int j = 0; j < 4; ++j)
        *(bf16x8*)&Bs[nr * 64 + c + 8 * j] = *(const bf16x8*)(Br + 8 * j);
#endif
    } else if (BF32) {
      const int kr = t >> 1, c = (t & 1) << 5;   // src row = k, cols = n
      const float* Bf = (const float*)Bp + zB + (long)(k0 + kr) * ldb + n0 + c;
#pragma unroll
      for (int j = 0; j < 32; ++j)
        Bs[(c + j) * 64 + kr] = (__bf16)(Bf[j] * bscale);
    } else {
      const int kr = t >> 1, c = (t & 1) << 5;
      const __bf16* Bb = (const __bf16*)Bp + zB + (long)(k0 + kr) * ldb + n0 + c;
#pragma unroll
      for (int j = 0; j < 32; ++j)
        Bs[(c + j) * 64 + kr] = Bb[j];
    }
#ifdef HAVE_TDM
    if (tdm_used && wv == 0) __builtin_amdgcn_s_wait_tensorcnt(0);
#else
    (void)tdm_used;
#endif
    __syncthreads();

    const __bf16* strip = &As[wv * 16 * 64];
#pragma unroll
    for (int kh = 0; kh < 64; kh += 32) {
      const bf16x16 a = frag_a_ld(strip, kh);
#pragma unroll
      for (int nf = 0; nf < 4; ++nf) {
        const bf16x16 b = frag_b_ld(Bs, nf * 16, kh);
        acc[nf] = __builtin_amdgcn_wmma_f32_16x16x32_bf16(
            false, a, false, b, (short)0, acc[nf], false, false);
      }
    }
    __syncthreads();
  }

  // ---- write out (C/D layout: VGPR r -> M = r + 8*half, N = lane&15) ----
  const int lane = t & 31;
  const int hf = lane >> 4;
  const int nn = lane & 15;
#pragma unroll
  for (int nf = 0; nf < 4; ++nf) {
#pragma unroll
    for (int r = 0; r < 8; ++r) {
      const int m = m0 + wv * 16 + r + (hf << 3);
      const int n = n0 + nf * 16 + nn;
      const float v = acc[nf][r];
      if (MODE == 0) {
        const int bb = m >> 10, s = m & 1023, h = n >> 6, d = n & 63;
        ((__bf16*)Op)[(((long)(bb * NH + h)) * NS + s) * ND + d] = (__bf16)v;
      } else if (MODE == 1) {
        ((__bf16*)Op)[((long)z << 20) + ((long)m << 10) + n] = (__bf16)v;
      } else if (MODE == 2) {
        const int bb = z / NH, h = z % NH;
        ((__bf16*)Op)[(((long)bb * NS + m) * NH + h) * ND + n] = (__bf16)v;
      } else {
        ((float*)Op)[(long)m * NC + n] = v;
      }
    }
  }
}

// ---------------------------------------------------------------------------
// Per-(b,q): pre-softmax head mix (W_pre), softmax over k, post mix (W_post).
// 12 waves (384 threads). LDS: 12x1024 bf16 + 12x1024 f32 (~73 KB).
// ---------------------------------------------------------------------------
__global__ __launch_bounds__(384)
void softmax_mix(const __bf16* __restrict__ S, const float* __restrict__ Wpre,
                 const float* __restrict__ Wpost, __bf16* __restrict__ P)
{
  __shared__ __align__(16) __bf16 sl[NH][NS];
  __shared__ __align__(16) float  pl[NH][NS];
  __shared__ float wpre[NH * NH];
  __shared__ float wpost[NH * NH];

  const int t = threadIdx.x;
  const int b = blockIdx.x >> 10;
  const int q = blockIdx.x & 1023;
  if (t < NH * NH) { wpre[t] = Wpre[t]; wpost[t] = Wpost[t]; }

  const long rowblk = (long)NS * NS;
  const __bf16* Sb = S + (long)b * NH * rowblk + (long)q * NS;
  for (int c = t; c < NH * (NS / 8); c += 384) {
    const int h = c >> 7;
    const int kc = (c & 127) << 3;
    *(bf16x8*)&sl[h][kc] = *(const bf16x8*)(Sb + (long)h * rowblk + kc);
  }
  __syncthreads();

  const int wv = t >> 5, lane = t & 31;
  if (wv < NH) {
    const int i = wv;
    float mx = -3.0e38f;
    for (int k = lane; k < NS; k += 32) {
      float a = 0.f;
#pragma unroll
      for (int h = 0; h < NH; ++h) a = fmaf(wpre[h * NH + i], (float)sl[h][k], a);
      pl[i][k] = a;
      mx = fmaxf(mx, a);
    }
#pragma unroll
    for (int off = 16; off >= 1; off >>= 1) mx = fmaxf(mx, __shfl_xor(mx, off, 32));
    float sum = 0.f;
    for (int k = lane; k < NS; k += 32) {
      const float e = __expf(pl[i][k] - mx);
      pl[i][k] = e;
      sum += e;
    }
#pragma unroll
    for (int off = 16; off >= 1; off >>= 1) sum += __shfl_xor(sum, off, 32);
    const float inv = 1.0f / sum;
    for (int k = lane; k < NS; k += 32) pl[i][k] *= inv;
  }
  __syncthreads();

  if (wv < NH) {
    const int h = wv;
    __bf16* Pr = P + (long)(b * NH + h) * rowblk + (long)q * NS;
    for (int k = lane; k < NS; k += 32) {
      float a = 0.f;
#pragma unroll
      for (int i = 0; i < NH; ++i) a = fmaf(wpost[i * NH + h], pl[i][k], a);
      Pr[k] = (__bf16)a;
    }
  }
}

// ---------------------------------------------------------------------------
extern "C" void kernel_launch(void* const* d_in, const int* in_sizes, int n_in,
                              void* d_out, int out_size, void* d_ws, size_t ws_size,
                              hipStream_t stream) {
  (void)in_sizes; (void)n_in; (void)out_size; (void)ws_size;
  const float* inq   = (const float*)d_in[0];
  const float* inkv  = (const float*)d_in[1];
  const float* Wq    = (const float*)d_in[2];
  const float* Wk    = (const float*)d_in[3];
  const float* Wv    = (const float*)d_in[4];
  const float* Wpre  = (const float*)d_in[5];
  const float* Wpost = (const float*)d_in[6];
  const float* Wo    = (const float*)d_in[7];
  float* out = (float*)d_out;

  const size_t qkvN = (size_t)NB * NH * NS * ND;   // 6,291,456
  const size_t attN = (size_t)NB * NH * NS * NS;   // 100,663,296
  __bf16* qb     = (__bf16*)d_ws;
  __bf16* kb     = qb + qkvN;
  __bf16* vb     = kb + qkvN;
  __bf16* Sb     = vb + qkvN;          // raw scores   (201 MB)
  __bf16* Pb     = Sb + attN;          // post-softmax (201 MB)
  __bf16* scores = Pb + attN;          // [B,S,H,D]

  const dim3 blk(128);

  // 1) QKV projections: [8192x768] x [768x768] -> bf16 [B,H,S,D]
  //    (q scale 1/sqrt(D)=0.125 folded into Wq staging)
  gemm_wmma<0, true, true, false><<<dim3(128, 12, 1), blk, 0, stream>>>(
      inq,  Wq, qb, NC, NC, NC / 64, 0, 0, 0.125f);
  gemm_wmma<0, true, true, false><<<dim3(128, 12, 1), blk, 0, stream>>>(
      inkv, Wk, kb, NC, NC, NC / 64, 0, 0, 1.0f);
  gemm_wmma<0, true, true, false><<<dim3(128, 12, 1), blk, 0, stream>>>(
      inkv, Wv, vb, NC, NC, NC / 64, 0, 0, 1.0f);

  // 2) raw scores: per (b,h): q[1024x64] x k^T -> bf16 [z][q][k]  (TDM staged)
  gemm_wmma<1, false, false, true><<<dim3(16, 16, NB * NH), blk, 0, stream>>>(
      qb, kb, Sb, ND, ND, ND / 64, (long)NS * ND, (long)NS * ND, 1.0f);

  // 3) pre-mix + softmax + post-mix, one block per (b,q)
  softmax_mix<<<dim3(NB * NS), dim3(384), 0, stream>>>(Sb, Wpre, Wpost, Pb);

  // 4) attn @ V: per (b,h): P[1024x1024] x v[1024x64] -> bf16 [B,S,H,D]
  gemm_wmma<2, false, false, false><<<dim3(16, 1, NB * NH), blk, 0, stream>>>(
      Pb, vb, scores, NS, ND, NS / 64, (long)NS * NS, (long)NS * ND, 1.0f);

  // 5) output projection: [8192x768] x Wo[768x768] -> f32 d_out
  gemm_wmma<3, false, true, false><<<dim3(128, 12, 1), blk, 0, stream>>>(
      scores, Wo, out, NC, NC, NC / 64, 0, 0, 1.0f);
}
